// EventDrivenODECell_3427383902461
// MI455X (gfx1250) — matrix operand order repaired and checked by reading, not compiled, not run on hardware
//
#include <hip/hip_runtime.h>
#include <hip/hip_bf16.h>
#include <math.h>

// ---------------------------------------------------------------------------
// EventDrivenODECell for MI455X (gfx1250).
// Compute-bound (~1000 FLOP/byte) -> bf16 WMMA 16x16x32, f32 accumulation.
// One wave owns a 16-row batch tile through the whole GEMM chain.
// v3: same structure as v2 (A-fragments register-resident per layer, single
// per-wave LDS X buffer, fast transcendentals) but the 16-iteration N-tile
// loops are explicitly NOT unrolled: v2's full unroll made the compiler
// preload all B-fragments, exceed 256 VGPRs (vgpr-msb mode) and spill them
// to scratch. Rolled N-loops keep live state ~150 VGPRs: A-fragments (<=80)
// + one accumulator + in-flight B loads, so B comes straight from
// global_load_b128 (L2-resident packed weights), no scratch traffic.
// ---------------------------------------------------------------------------

#define HID 256
#define EVT 64
#define NSTEPS 5
#define XSTR 320           // halfs per LDS activation row (HID + EVT)
#define WAVES_PER_BLOCK 4
#define ROWS_PER_WAVE 16
#define LDS_PER_WAVE (XSTR*16*2 /*X*/ + 256*16*4 /*hbuf*/)   // 26624 B

typedef __attribute__((ext_vector_type(16))) __bf16 v16bf;
typedef __attribute__((ext_vector_type(8)))  float  v8f;

static __device__ __forceinline__ unsigned short f2bf(float x) {
    // round-to-nearest (ties away) bf16 conversion: 2 VALU ops
    return (unsigned short)((__float_as_uint(x) + 0x8000u) >> 16);
}
static __device__ __forceinline__ float fast_tanh(float x) {
    // 1 - 2/(e^{2x}+1); saturates correctly to +-1 for large |x|
    float e = __expf(2.0f * x);
    return 1.0f - 2.0f * __builtin_amdgcn_rcpf(e + 1.0f);
}
static __device__ __forceinline__ float fast_sigmoid(float x) {
    return __builtin_amdgcn_rcpf(1.0f + __expf(-x));
}

// Preload the wave's full 16xK activation tile as WMMA A-fragments.
// A layout (16-bit A 16x32): lane<16 holds M=lane, halfs {K0..7, K16..23};
// lane>=16 holds M=lane-16, halfs {K8..15, K24..31}.
template<int KT>
static __device__ __forceinline__ void load_afrags(const unsigned short* Xlds,
                                                   int colbaseBytes, int lane,
                                                   uint4* A) {
    const char* rowp = (const char*)Xlds + (lane & 15) * (XSTR * 2)
                       + colbaseBytes + ((lane >> 4) ? 16 : 0);
#pragma unroll
    for (int k = 0; k < KT; ++k) {
        A[2 * k]     = *(const uint4*)(rowp + 64 * k);        // ds_load_b128
        A[2 * k + 1] = *(const uint4*)(rowp + 64 * k + 32);   // ds_load_b128
    }
}

// One 16x16 output tile: C = bias + A(16,KT*32) x B(KT*32,16).
// wfrag: global bf16 weights in B-fragment order: frag(n,k) = 32 lanes x 32B.
template<int KT>
static __device__ __forceinline__ v8f gemm_acc(const uint4* A,
                                               const unsigned short* __restrict__ wfrag,
                                               int nTile, float bias, int lane) {
    v8f acc = {bias, bias, bias, bias, bias, bias, bias, bias};
    const uint4* wp = (const uint4*)wfrag + (size_t)nTile * KT * 64 + lane * 2;
#pragma unroll
    for (int k = 0; k < KT; ++k) {
        union { uint4 u[2]; v16bf v; } Am, Bm;
        Am.u[0] = A[2 * k];
        Am.u[1] = A[2 * k + 1];
        Bm.u[0] = wp[0];                                      // global_load_b128
        Bm.u[1] = wp[1];                                      // global_load_b128
        wp += 64;
        acc = __builtin_amdgcn_wmma_f32_16x16x32_bf16(
            false, Am.v, false, Bm.v, (short)0, acc, false, false);
    }
    return acc;
}

// Prologue: repack row-major fp32 weight W[N][IN] into bf16 WMMA B-fragment
// order. B layout (16-bit B 32x16): lane<16 -> N=lane, K={0..7, 16..23};
// lane>=16 -> N=lane-16, K={8..15, 24..31}.
extern "C" __global__ void prep_weights_kernel(const float* __restrict__ W,
                                               unsigned short* __restrict__ dst,
                                               int Kt, int IN, int total) {
    int idx = blockIdx.x * blockDim.x + threadIdx.x;
    if (idx >= total) return;
    int j    = idx & 15;
    int lane = (idx >> 4) & 31;
    int frag = idx >> 9;
    int n = frag / Kt;
    int k = frag - n * Kt;
    int row = n * 16 + (lane & 15);             // output-feature index (N)
    int col = k * 32 + ((lane >> 4) * 8) + ((j >> 3) * 16) + (j & 7);  // input-feature index (K)
    dst[idx] = f2bf(W[(size_t)row * IN + col]);
}

extern "C" __global__ void __launch_bounds__(WAVES_PER_BLOCK * 32, 1)
EventDrivenODECell_3427383902461_kernel(
    const float* __restrict__ h_prev, const float* __restrict__ event,
    const float* __restrict__ t_start_p, const float* __restrict__ t_end_p,
    const float* __restrict__ Wt1, const float* __restrict__ bt1,
    const float* __restrict__ Wt2, const float* __restrict__ bt2,
    const float* __restrict__ bd1, const float* __restrict__ bd2,
    const float* __restrict__ bd3, const float* __restrict__ be1,
    const float* __restrict__ be2, const float* __restrict__ bg,
    const unsigned short* __restrict__ wd1f, const unsigned short* __restrict__ wd2f,
    const unsigned short* __restrict__ wd3f, const unsigned short* __restrict__ we1f,
    const unsigned short* __restrict__ we2f, const unsigned short* __restrict__ wgf,
    float* __restrict__ out) {
    extern __shared__ char smem[];
    const int tid  = threadIdx.x;
    const int lane = tid & 31;
    const int wid  = tid >> 5;
    const int lc   = lane & 15;
    const int hi   = lane >> 4;

    char* wbase = smem + (size_t)wid * LDS_PER_WAVE;
    unsigned short* X    = (unsigned short*)wbase;
    float*          hbuf = (float*)(wbase + XSTR * 16 * 2);

    const int m0 = blockIdx.x * (WAVES_PER_BLOCK * ROWS_PER_WAVE) + wid * ROWS_PER_WAVE;

    // ---- init: h_prev -> hbuf (f32) and X (bf16) ----
    #pragma clang loop unroll(disable)
    for (int idx = lane; idx < 16 * 256; idx += 32) {
        int r = idx >> 8, c = idx & 255;
        float v = h_prev[(size_t)(m0 + r) * HID + c];
        hbuf[r * 256 + c] = v;
        X[r * XSTR + c]   = f2bf(v);
    }

    const float t0  = *t_start_p;
    const float dtv = (*t_end_p - t0) * (1.0f / NSTEPS);

    uint4 A[20];  // A-fragment registers: up to KT=10 (2 uint4 per K-tile)

    #pragma clang loop unroll(disable)
    for (int s = 0; s < NSTEPS; ++s) {
        float t = t0 + (float)s * dtv;
        // time embedding (identical for every row): lane j owns te[j]
        float u  = fast_tanh(t * Wt1[lane] + bt1[lane]);
        float te = bt2[lane];
        #pragma unroll 8
        for (int i = 0; i < 32; ++i) te += Wt2[lane * 32 + i] * __shfl(u, i, 32);
        unsigned short teb = f2bf(te);
        #pragma unroll
        for (int r = 0; r < 16; ++r) X[r * XSTR + 256 + lane] = teb;

        // z1 = tanh([h,te] @ Wd1^T + bd1), K=288: preload A, overwrite X
        load_afrags<9>(X, 0, lane, A);
        #pragma clang loop unroll(disable)
        for (int n = 0; n < 16; ++n) {
            v8f acc = gemm_acc<9>(A, wd1f, n, bd1[n * 16 + lc], lane);
            unsigned short* dst = X + (hi * 8) * XSTR + n * 16 + lc;
            #pragma unroll
            for (int r = 0; r < 8; ++r) dst[r * XSTR] = f2bf(fast_tanh(acc[r]));
        }
        // z2 = tanh(z1 @ Wd2^T + bd2), K=256
        load_afrags<8>(X, 0, lane, A);
        #pragma clang loop unroll(disable)
        for (int n = 0; n < 16; ++n) {
            v8f acc = gemm_acc<8>(A, wd2f, n, bd2[n * 16 + lc], lane);
            unsigned short* dst = X + (hi * 8) * XSTR + n * 16 + lc;
            #pragma unroll
            for (int r = 0; r < 8; ++r) dst[r * XSTR] = f2bf(fast_tanh(acc[r]));
        }
        // h += dt*(z2 @ Wd3^T + bd3): update hbuf(f32), write new h bf16 to X
        load_afrags<8>(X, 0, lane, A);
        #pragma clang loop unroll(disable)
        for (int n = 0; n < 16; ++n) {
            v8f acc = gemm_acc<8>(A, wd3f, n, bd3[n * 16 + lc], lane);
            float* hp = hbuf + (hi * 8) * 256 + n * 16 + lc;
            unsigned short* dst = X + (hi * 8) * XSTR + n * 16 + lc;
            #pragma unroll
            for (int r = 0; r < 8; ++r) {
                float hn = hp[r * 256] + dtv * acc[r];
                hp[r * 256] = hn;
                dst[r * XSTR] = f2bf(hn);
            }
        }
    }

    // ---- event phase: X cols 0..255 = final h bf16; add event cols 256..319
    #pragma unroll
    for (int r = 0; r < 16; ++r) {
        size_t eb = (size_t)(m0 + r) * EVT;
        X[r * XSTR + 256 + lane] = f2bf(event[eb + lane]);
        X[r * XSTR + 288 + lane] = f2bf(event[eb + 32 + lane]);
    }
    // z = relu([h,e] @ We1^T + be1), K=320 -> overwrite X cols 0..255
    load_afrags<10>(X, 0, lane, A);
    #pragma clang loop unroll(disable)
    for (int n = 0; n < 16; ++n) {
        v8f acc = gemm_acc<10>(A, we1f, n, be1[n * 16 + lc], lane);
        unsigned short* dst = X + (hi * 8) * XSTR + n * 16 + lc;
        #pragma unroll
        for (int r = 0; r < 8; ++r) dst[r * XSTR] = f2bf(fmaxf(acc[r], 0.0f));
    }
    // upd = z @ We2^T + be2 ; gate = sigmoid(e @ Wg^T + bg) ; out = h + gate*upd
    uint4 Ag[4];
    load_afrags<8>(X, 0,   lane, A);   // z (cols 0..255)
    load_afrags<2>(X, 512, lane, Ag);  // event (cols 256..319)
    #pragma clang loop unroll(disable)
    for (int n = 0; n < 16; ++n) {
        v8f upd = gemm_acc<8>(A,  we2f, n, be2[n * 16 + lc], lane);
        v8f gat = gemm_acc<2>(Ag, wgf,  n, bg [n * 16 + lc], lane);
        const float* hp = hbuf + (hi * 8) * 256 + n * 16 + lc;
        float* op = out + (size_t)(m0 + hi * 8) * HID + n * 16 + lc;
        #pragma unroll
        for (int r = 0; r < 8; ++r) {
            op[(size_t)r * HID] = hp[r * 256] + fast_sigmoid(gat[r]) * upd[r];
        }
    }
}

// ws layout (bf16 halfs)
#define OFF_D1 0
#define OFF_D2 (OFF_D1 + 16*9*512)    // 73728
#define OFF_D3 (OFF_D2 + 16*8*512)    // +65536
#define OFF_E1 (OFF_D3 + 16*8*512)    // +65536
#define OFF_E2 (OFF_E1 + 16*10*512)   // +81920
#define OFF_G  (OFF_E2 + 16*8*512)    // +65536

extern "C" void kernel_launch(void* const* d_in, const int* in_sizes, int n_in,
                              void* d_out, int out_size, void* d_ws, size_t ws_size,
                              hipStream_t stream) {
    const float* h_prev = (const float*)d_in[0];
    const float* event  = (const float*)d_in[1];
    const float* t_s    = (const float*)d_in[2];
    const float* t_e    = (const float*)d_in[3];
    const float* Wt1    = (const float*)d_in[4];
    const float* bt1    = (const float*)d_in[5];
    const float* Wt2    = (const float*)d_in[6];
    const float* bt2    = (const float*)d_in[7];
    const float* Wd1    = (const float*)d_in[8];
    const float* bd1    = (const float*)d_in[9];
    const float* Wd2    = (const float*)d_in[10];
    const float* bd2    = (const float*)d_in[11];
    const float* Wd3    = (const float*)d_in[12];
    const float* bd3    = (const float*)d_in[13];
    const float* We1    = (const float*)d_in[14];
    const float* be1    = (const float*)d_in[15];
    const float* We2    = (const float*)d_in[16];
    const float* be2    = (const float*)d_in[17];
    const float* Wg     = (const float*)d_in[18];
    const float* bg     = (const float*)d_in[19];
    unsigned short* ws  = (unsigned short*)d_ws;

    // pre-pack all weights into WMMA B-fragment bf16 order
    struct { const float* W; int off; int Kt; int IN; int total; } mats[6] = {
        { Wd1, OFF_D1, 9,  288, 16*9*512  },
        { Wd2, OFF_D2, 8,  256, 16*8*512  },
        { Wd3, OFF_D3, 8,  256, 16*8*512  },
        { We1, OFF_E1, 10, 320, 16*10*512 },
        { We2, OFF_E2, 8,  256, 16*8*512  },
        { Wg,  OFF_G,  2,  64,  16*2*512  },
    };
    for (int i = 0; i < 6; ++i) {
        int tot = mats[i].total;
        prep_weights_kernel<<<(tot + 255) / 256, 256, 0, stream>>>(
            mats[i].W, ws + mats[i].off, mats[i].Kt, mats[i].IN, tot);
    }

    int rows = in_sizes[0] / HID;                        // 65536
    int grid = rows / (WAVES_PER_BLOCK * ROWS_PER_WAVE); // 1024 blocks of 64 rows
    size_t lds = (size_t)WAVES_PER_BLOCK * LDS_PER_WAVE; // 106496 bytes
    EventDrivenODECell_3427383902461_kernel<<<grid, WAVES_PER_BLOCK * 32, lds, stream>>>(
        h_prev, event, t_s, t_e, Wt1, bt1, Wt2, bt2,
        bd1, bd2, bd3, be1, be2, bg,
        ws + OFF_D1, ws + OFF_D2, ws + OFF_D3, ws + OFF_E1, ws + OFF_E2, ws + OFF_G,
        (float*)d_out);
}
